// GNN_32693291057798
// MI455X (gfx1250) — matrix operand order repaired and checked by reading, not compiled, not run on hardware
//
#include <hip/hip_runtime.h>
#include <hip/hip_bf16.h>

// ---------------------------------------------------------------------------
// GNN (3-layer GraphConv) for MI455X / gfx1250, wave32 + WMMA bf16.
// Pipeline: pack weights -> [zero, scatter-atomics, WMMA GEMM] x2 layers -> [zero, scatter-atomics, VALU FMA] layer 3.
// ---------------------------------------------------------------------------

typedef __attribute__((ext_vector_type(16))) __bf16 v16bf;
typedef __attribute__((ext_vector_type(8)))  float  v8f;

#define NN 100000   // nodes
#define EE 400000   // edges

// ---------------------------------------------------------------------------
__global__ void zero_f32x4(float4* __restrict__ p, long long n4) {
  long long i = (long long)blockIdx.x * blockDim.x + threadIdx.x;
  if (i < n4) p[i] = make_float4(0.f, 0.f, 0.f, 0.f);
}

// agg[dst, :] += w_e * feat[src, :]; 4 dims per thread (b128 gather loads,
// 4x global_atomic_add_f32). Wave-uniform edge metadata (32 lanes = 128 dims).
__global__ void scatter_add4(const float* __restrict__ feat,
                             const int*   __restrict__ ei,   // [2,E]
                             const float* __restrict__ ew,
                             float* __restrict__ agg,
                             int E, int dimBits) {
  const int qBits = dimBits - 2;                  // log2(D/4)
  long long idx = (long long)blockIdx.x * blockDim.x + threadIdx.x;
  if (idx >= ((long long)E << qBits)) return;
  int e  = (int)(idx >> qBits);
  int d4 = (int)(idx & ((1 << qBits) - 1));
  int s  = ei[e];
  int t  = ei[E + e];
  float w = ew[e];
  const float4 v = *(const float4*)&feat[((size_t)s << dimBits) + (d4 << 2)];
  float* dst = &agg[((size_t)t << dimBits) + (d4 << 2)];
  atomicAdd(dst + 0, w * v.x);
  atomicAdd(dst + 1, w * v.y);
  atomicAdd(dst + 2, w * v.z);
  atomicAdd(dst + 3, w * v.w);
}

// Pack W[K][Dout] (f32 row-major) into bf16 WMMA B-fragments:
//   Wp[ ((ntile*Ksteps + ks)*32 + lane)*16 + e ]
// lanes 0-15: col N = lane, K = ks*32 + {0..7, 16..23}; lanes 16-31: col N = lane-16,
// K = ks*32 + {8..15, 24..31}. Each lane's fragment is one contiguous 32-byte chunk.
__global__ void pack_w_bf16(const float* __restrict__ W, __bf16* __restrict__ Wp,
                            int K, int Dout) {
  int idx = blockIdx.x * blockDim.x + threadIdx.x;
  if (idx >= K * Dout) return;
  int e      = idx & 15;
  int lane   = (idx >> 4) & 31;
  int rest   = idx >> 9;
  int Ksteps = K >> 5;
  int ks     = rest % Ksteps;
  int ntile  = rest / Ksteps;
  int n = (ntile << 4) + (lane & 15);
  int k = (ks << 5) + ((lane >> 4) << 3) + (e & 7) + ((e >> 3) << 4);
  Wp[idx] = (__bf16)W[(size_t)k * Dout + n];
}

__device__ __forceinline__ v16bf cvt16(float4 a0, float4 a1, float4 a2, float4 a3) {
  union { v16bf v; __bf16 h[16]; } af;
  af.h[0]  = (__bf16)a0.x; af.h[1]  = (__bf16)a0.y;
  af.h[2]  = (__bf16)a0.z; af.h[3]  = (__bf16)a0.w;
  af.h[4]  = (__bf16)a1.x; af.h[5]  = (__bf16)a1.y;
  af.h[6]  = (__bf16)a1.z; af.h[7]  = (__bf16)a1.w;
  af.h[8]  = (__bf16)a2.x; af.h[9]  = (__bf16)a2.y;
  af.h[10] = (__bf16)a2.z; af.h[11] = (__bf16)a2.w;
  af.h[12] = (__bf16)a3.x; af.h[13] = (__bf16)a3.y;
  af.h[14] = (__bf16)a3.z; af.h[15] = (__bf16)a3.w;
  return af.v;
}

// Fused GEMM: out = act( Arel @ WrelP + Aroot @ WrootP + bias ).
// Block = 256 threads = 8 waves; each wave owns NTILES independent 16x16 C
// tiles (block tile = 16 rows x 128*NTILES cols, gridDim.y == 1).
// Barrier-free; each lane gathers its A fragment straight from global in WMMA
// layout (redundancy across the 8 waves is absorbed by the shared WGP$).
// Manually software-pipelined (distance 1, unroll x2, disjoint X/Y register
// sets) so every load retires at least half an iteration after issue and no
// register-rotation movs / WAR nops are needed.
template <int NTILES>
__global__ void __launch_bounds__(256)
gemm_fused_bf16(const float*  __restrict__ Arel,
                const float*  __restrict__ Aroot,
                const __bf16* __restrict__ WrelP,
                const __bf16* __restrict__ WrootP,
                const float*  __restrict__ bias,
                float* __restrict__ out,
                int K, int Dout, int doRelu) {
  const int lane      = threadIdx.x & 31;
  const int wave      = threadIdx.x >> 5;
  const int rowBase   = blockIdx.x << 4;
  const int ntileBase = wave * NTILES;
  const int Ksteps    = K >> 5;          // even for all layers (32, 8)

  // A-fragment geometry: lane -> row m = lane%16, half = lane/16.
  // frag elems 0..7  = K: ks*32 + half*8 + {0..7}
  // frag elems 8..15 = K: ks*32 + 16 + half*8 + {0..7}
  const int m    = lane & 15;
  const int half = lane >> 4;
  const size_t rowOff = (size_t)(rowBase + m) * K + (half << 3);

  v8f acc0 = (v8f){0.f, 0.f, 0.f, 0.f, 0.f, 0.f, 0.f, 0.f};
  v8f acc1 = acc0;

  for (int phase = 0; phase < 2; ++phase) {
    const float*  __restrict__ A  = phase ? Aroot  : Arel;
    const __bf16* __restrict__ Wp = phase ? WrootP : WrelP;

    const float* aRow = A + rowOff;                      // +32 floats / step
    const size_t tileStride = (size_t)Ksteps << 9;       // 512 bf16 / step
    const __bf16* b0Row = Wp + (size_t)ntileBase * tileStride
                             + ((size_t)lane << 4);
    const __bf16* b1Row = (NTILES > 1) ? (b0Row + tileStride) : b0Row;

    // ---- prologue: step 0 into X set ----
    float4 xA0 = *(const float4*)(aRow + 0);
    float4 xA1 = *(const float4*)(aRow + 4);
    float4 xA2 = *(const float4*)(aRow + 16);
    float4 xA3 = *(const float4*)(aRow + 20);
    v16bf  xB0 = *(const v16bf*)(b0Row);
    v16bf  xB1 = (NTILES > 1) ? *(const v16bf*)(b1Row) : xB0;

    for (int ks = 0; ks < Ksteps; ks += 2) {
      // issue loads for odd step ks+1 (Y set)
      float4 yA0 = *(const float4*)(aRow + 32);
      float4 yA1 = *(const float4*)(aRow + 36);
      float4 yA2 = *(const float4*)(aRow + 48);
      float4 yA3 = *(const float4*)(aRow + 52);
      v16bf  yB0 = *(const v16bf*)(b0Row + 512);
      v16bf  yB1 = (NTILES > 1) ? *(const v16bf*)(b1Row + 512) : yB0;

      // consume even step ks (X set, loaded >= 1 iteration ago)
      {
        v16bf af = cvt16(xA0, xA1, xA2, xA3);
        acc0 = __builtin_amdgcn_wmma_f32_16x16x32_bf16(
                   false, af, false, xB0, (short)0, acc0, false, false);
        if (NTILES > 1)
          acc1 = __builtin_amdgcn_wmma_f32_16x16x32_bf16(
                     false, af, false, xB1, (short)0, acc1, false, false);
      }

      // issue loads for next even step ks+2 into X set (uniform-guarded tail)
      if (ks + 2 < Ksteps) {
        xA0 = *(const float4*)(aRow + 64);
        xA1 = *(const float4*)(aRow + 68);
        xA2 = *(const float4*)(aRow + 80);
        xA3 = *(const float4*)(aRow + 84);
        xB0 = *(const v16bf*)(b0Row + 1024);
        if (NTILES > 1) xB1 = *(const v16bf*)(b1Row + 1024);
      }
      aRow  += 64;
      b0Row += 1024;
      b1Row += 1024;

      // consume odd step ks+1 (Y set)
      {
        v16bf af = cvt16(yA0, yA1, yA2, yA3);
        acc0 = __builtin_amdgcn_wmma_f32_16x16x32_bf16(
                   false, af, false, yB0, (short)0, acc0, false, false);
        if (NTILES > 1)
          acc1 = __builtin_amdgcn_wmma_f32_16x16x32_bf16(
                     false, af, false, yB1, (short)0, acc1, false, false);
      }
    }
  }

  // C layout: VGPR r holds M = r (lanes 0-15) / M = 8+r (lanes 16-31), N = lane%16.
  const int nl = lane & 15;
  const int mo = (lane >> 4) << 3;
#pragma unroll
  for (int t = 0; t < NTILES; ++t) {
    const v8f accv = (t == 0) ? acc0 : acc1;
    const int col = ((ntileBase + t) << 4) + nl;
    const float bv = bias[col];
#pragma unroll
    for (int r = 0; r < 8; ++r) {
      float v = accv[r] + bv;
      if (doRelu) v = fmaxf(v, 0.0f);
      out[(size_t)(rowBase + mo + r) * Dout + col] = v;
    }
  }
}

// Layer 3: Dout=4 too narrow for WMMA -> VALU FMA kernel (tiny weights, L2-hot).
__global__ void layer3_small(const float* __restrict__ agg,
                             const float* __restrict__ h,
                             const float* __restrict__ Wrel,   // [128,4]
                             const float* __restrict__ brel,   // [4]
                             const float* __restrict__ Wroot,  // [128,4]
                             float* __restrict__ out, int N) {
  int idx = blockIdx.x * blockDim.x + threadIdx.x;
  if (idx >= N * 4) return;
  int n = idx >> 2, j = idx & 3;
  const float* ar = agg + (size_t)n * 128;
  const float* hr = h   + (size_t)n * 128;
  float s = brel[j];
#pragma unroll 4
  for (int k = 0; k < 128; ++k)
    s = fmaf(ar[k], Wrel[k * 4 + j], fmaf(hr[k], Wroot[k * 4 + j], s));
  out[idx] = s;
}

// ---------------------------------------------------------------------------
extern "C" void kernel_launch(void* const* d_in, const int* in_sizes, int n_in,
                              void* d_out, int out_size, void* d_ws, size_t ws_size,
                              hipStream_t stream) {
  (void)in_sizes; (void)n_in; (void)out_size; (void)ws_size;

  const float* x      = (const float*)d_in[0];
  const int*   ei     = (const int*)  d_in[1];   // [2,E]
  const float* ew     = (const float*)d_in[2];
  // d_in[3] = batch (unused)
  const float* Wrel1  = (const float*)d_in[4];
  const float* brel1  = (const float*)d_in[5];
  const float* Wroot1 = (const float*)d_in[6];
  const float* Wrel2  = (const float*)d_in[7];
  const float* brel2  = (const float*)d_in[8];
  const float* Wroot2 = (const float*)d_in[9];
  const float* Wrel3  = (const float*)d_in[10];
  const float* brel3  = (const float*)d_in[11];
  const float* Wroot3 = (const float*)d_in[12];
  float* out = (float*)d_out;

  const int N = NN, E = EE;

  // Workspace carve-up (agg buffer reused across layers).
  float*  agg = (float*)d_ws;                           // N*1024 f32 (max)
  float*  h1  = agg + (size_t)N * 1024;                 // N*256 f32
  float*  h2  = h1  + (size_t)N * 256;                  // N*128 f32
  __bf16* wpRel1  = (__bf16*)(h2 + (size_t)N * 128);    // 1024*256
  __bf16* wpRoot1 = wpRel1  + 1024 * 256;               // 1024*256
  __bf16* wpRel2  = wpRoot1 + 1024 * 256;               // 256*128
  __bf16* wpRoot2 = wpRel2  + 256 * 128;                // 256*128

  // Pack weights into bf16 WMMA B-fragment layout.
  pack_w_bf16<<<(1024 * 256 + 255) / 256, 256, 0, stream>>>(Wrel1,  wpRel1,  1024, 256);
  pack_w_bf16<<<(1024 * 256 + 255) / 256, 256, 0, stream>>>(Wroot1, wpRoot1, 1024, 256);
  pack_w_bf16<<<( 256 * 128 + 255) / 256, 256, 0, stream>>>(Wrel2,  wpRel2,  256, 128);
  pack_w_bf16<<<( 256 * 128 + 255) / 256, 256, 0, stream>>>(Wroot2, wpRoot2, 256, 128);

  // ---- Layer 1: 1024 -> 256 --------------------------------------------
  {
    long long n4 = (long long)N * 1024 / 4;
    zero_f32x4<<<(unsigned)((n4 + 255) / 256), 256, 0, stream>>>((float4*)agg, n4);
    long long w = (long long)E * (1024 / 4);
    scatter_add4<<<(unsigned)((w + 255) / 256), 256, 0, stream>>>(x, ei, ew, agg, E, 10);
    gemm_fused_bf16<2><<<dim3(N / 16, 1), 256, 0, stream>>>(
        agg, x, wpRel1, wpRoot1, brel1, h1, 1024, 256, 1);
  }

  // ---- Layer 2: 256 -> 128 ---------------------------------------------
  {
    long long n4 = (long long)N * 256 / 4;
    zero_f32x4<<<(unsigned)((n4 + 255) / 256), 256, 0, stream>>>((float4*)agg, n4);
    long long w = (long long)E * (256 / 4);
    scatter_add4<<<(unsigned)((w + 255) / 256), 256, 0, stream>>>(h1, ei, ew, agg, E, 8);
    gemm_fused_bf16<1><<<dim3(N / 16, 1), 256, 0, stream>>>(
        agg, h1, wpRel2, wpRoot2, brel2, h2, 256, 128, 1);
  }

  // ---- Layer 3: 128 -> 4 (no relu) -------------------------------------
  {
    long long n4 = (long long)N * 128 / 4;
    zero_f32x4<<<(unsigned)((n4 + 255) / 256), 256, 0, stream>>>((float4*)agg, n4);
    long long w = (long long)E * (128 / 4);
    scatter_add4<<<(unsigned)((w + 255) / 256), 256, 0, stream>>>(h2, ei, ew, agg, E, 7);
    layer3_small<<<(N * 4 + 255) / 256, 256, 0, stream>>>(
        agg, h2, Wrel3, brel3, Wroot3, out, N);
  }
}
